// GNNTransformer_33457795236067
// MI455X (gfx1250) — matrix operand order repaired and checked by reading, not compiled, not run on hardware
//
#include <hip/hip_runtime.h>
#include <hip/hip_bf16.h>
#include <math.h>

// ---------------------------------------------------------------------------
// Problem constants (from the reference)
// ---------------------------------------------------------------------------
#define BATCH    8
#define NMAX     2048
#define INFEAT   512
#define HID      128
#define NHEAD    4
#define NLAYER   3
#define KNN      8
#define GRAPH_DIM 32
#define ADJ_WORDS 64            // 2048 / 32 bit-words per adjacency row (fixed stride)

typedef __attribute__((ext_vector_type(16))) _Float16 v16h;
typedef __attribute__((ext_vector_type(8)))  float    v8f;

// ---------------------------------------------------------------------------
// Weight pre-pack: f32 block-layout W -> f16 per-column-contiguous wT[n][k].
// Source index follows the [(n>>7), K, 128] block view (== row-major [K,Nout]
// when Nout==128; == head-concat Ws[head,K,128] when Nout==512).
// After packing, a WMMA B-fragment (16 consecutive k for one column) is a
// single aligned 32-byte vector load.
// ---------------------------------------------------------------------------
__global__ __launch_bounds__(256)
void pack_w_kernel(const float* __restrict__ W, _Float16* __restrict__ wT,
                   int K, int total)
{
    const int t = blockIdx.x * 256 + threadIdx.x;
    if (t >= total) return;
    const int n = t / K, k = t % K;
    wT[(long)n * K + k] = (_Float16)W[((long)(n >> 7) * K + k) * 128 + (n & 127)];
}

static void pack_w(const float* W, _Float16* wT, int K, int Nout, hipStream_t stream)
{
    const int total = K * Nout;
    pack_w_kernel<<<(total + 255) / 256, 256, 0, stream>>>(W, wT, K, total);
}

// ---------------------------------------------------------------------------
// A-fragment loader: 16 f32 -> v16h in contiguous k order (same order as B).
// Lane half selects K-base 0/16; element e maps to K = base + e,
// i.e. one aligned 16-float run => four float4 loads (global_load_b128).
// ---------------------------------------------------------------------------
template<bool RIN>
__device__ __forceinline__ v16h load_a_frag(const float* __restrict__ base)
{
    const float4* ap = (const float4*)base;
    const float4 q0 = ap[0], q1 = ap[1], q2 = ap[2], q3 = ap[3];
    const float t[16] = {q0.x, q0.y, q0.z, q0.w,  q1.x, q1.y, q1.z, q1.w,
                         q2.x, q2.y, q2.z, q2.w,  q3.x, q3.y, q3.z, q3.w};
    v16h f;
#pragma unroll
    for (int e = 0; e < 16; ++e) {
        float v = t[e];
        if constexpr (RIN) v = fmaxf(v, 0.0f);
        f[e] = (_Float16)v;
    }
    return f;
}

// ---------------------------------------------------------------------------
// WMMA GEMM:  C[M,Nout] = epilogue( A[M,KT] @ Wt (+ A2 @ Wt2) + bias )
//   - KT compile-time (128 or 512): K-loop fully unrolled.
//   - Each wave owns a 64x16 strip (4 M-tiles): the B fragment is loaded once
//     per k-step (one 32B vector load from the packed f16 weights) and reused
//     by 4 back-to-back v_wmma ops.
//   - DUAL fuses h = xn@Ws + m@Wn into one accumulator pass (no C round-trip).
//   - All epilogue features are template flags -> straight-line code.
// ---------------------------------------------------------------------------
template<int KT, bool RIN, bool ROUT, bool HB, bool HS, bool HR, bool DUAL>
__global__ __launch_bounds__(128)
void wmma_gemm_kernel(const float* __restrict__ A, const float* __restrict__ A2,
                      const _Float16* __restrict__ Wt, const _Float16* __restrict__ Wt2,
                      const float* __restrict__ bias, const float* __restrict__ scale,
                      const float* __restrict__ shift, const float* __restrict__ residual,
                      float* __restrict__ C, int M, int Nout, float scale_mul)
{
    constexpr int MT = 4;                                 // 4 M-tiles per wave
    const int lane  = threadIdx.x;                        // 0..31
    const int wave  = threadIdx.y;                        // 0..3
    const int tileM = blockIdx.x * (16 * MT);
    const int tileN = (blockIdx.y * 4 + wave) * 16;

    const int  lhalf  = lane & 15;
    const int  hsel   = lane >> 4;
    const int  akbase = hsel << 4;                        // A: K-base 0 or 16
    const int  bkbase = hsel << 4;                        // B: K-base 0 or 16
    const int  colB   = tileN + lhalf;                    // B col supplied by lane

    const _Float16* wcol  = Wt  + (long)colB * KT + bkbase;
    const _Float16* wcol2 = DUAL ? (Wt2 + (long)colB * KT + bkbase) : nullptr;

    const int orow0 = tileM + (hsel << 3);                // C rows: +0..7 / +8..15
    const int ocol  = tileN + lhalf;

    v8f acc[MT];
#pragma unroll
    for (int mt = 0; mt < MT; ++mt)
#pragma unroll
        for (int r = 0; r < 8; ++r) acc[mt][r] = 0.0f;

#pragma unroll
    for (int k0 = 0; k0 < KT; k0 += 32) {
        const v16h bf = *(const v16h*)(wcol + k0);        // one 32B vector load
#pragma unroll
        for (int mt = 0; mt < MT; ++mt) {
            const v16h af = load_a_frag<RIN>(
                A + (long)(tileM + mt * 16 + lhalf) * KT + k0 + akbase);
            acc[mt] = __builtin_amdgcn_wmma_f32_16x16x32_f16(
                false, af, false, bf, (short)0, acc[mt], false, false);
        }
    }

    if constexpr (DUAL) {
#pragma unroll
        for (int k0 = 0; k0 < KT; k0 += 32) {
            const v16h bf = *(const v16h*)(wcol2 + k0);
#pragma unroll
            for (int mt = 0; mt < MT; ++mt) {
                const v16h af = load_a_frag<false>(
                    A2 + (long)(tileM + mt * 16 + lhalf) * KT + k0 + akbase);
                acc[mt] = __builtin_amdgcn_wmma_f32_16x16x32_f16(
                    false, af, false, bf, (short)0, acc[mt], false, false);
            }
        }
    }

    float bv = 0.0f, sv = 1.0f, shv = 0.0f;
    if constexpr (HB) bv = bias[ocol];
    if constexpr (HS) { sv = scale[ocol] * scale_mul; shv = shift[ocol]; }
    (void)bv; (void)sv; (void)shv; (void)scale_mul;

#pragma unroll
    for (int mt = 0; mt < MT; ++mt) {
#pragma unroll
        for (int r = 0; r < 8; ++r) {
            const long ri = (long)(orow0 + mt * 16 + r) * Nout + ocol;
            float v = acc[mt][r];
            if constexpr (HB) v += bv;
            if constexpr (HS) v = v * sv + shv;
            if constexpr (HR) v += residual[ri];
            if constexpr (ROUT) v = fmaxf(v, 0.0f);
            C[ri] = v;
        }
    }
}

template<int KT, bool RIN, bool ROUT, bool HB, bool HS, bool HR, bool DUAL>
static void launch_gemm(const float* A, const float* A2,
                        const _Float16* Wt, const _Float16* Wt2,
                        const float* bias, const float* scale, const float* shift,
                        const float* residual, float* C, int M, int Nout,
                        float scale_mul, hipStream_t stream)
{
    dim3 grid(M / 64, Nout / 64), block(32, 4);
    wmma_gemm_kernel<KT, RIN, ROUT, HB, HS, HR, DUAL>
        <<<grid, block, 0, stream>>>(A, A2, Wt, Wt2, bias, scale, shift, residual,
                                     C, M, Nout, scale_mul);
}

// ---------------------------------------------------------------------------
// LayerNorm over H=128 features, one row per 128-thread block.
// ---------------------------------------------------------------------------
__global__ __launch_bounds__(128)
void layernorm_kernel(const float* __restrict__ x, const float* __restrict__ g,
                      const float* __restrict__ be, float* __restrict__ out)
{
    __shared__ float red[128];
    const int row = blockIdx.x;
    const int d   = threadIdx.x;
    const float v = x[(long)row * HID + d];
    red[d] = v; __syncthreads();
    for (int s = 64; s > 0; s >>= 1) { if (d < s) red[d] += red[d + s]; __syncthreads(); }
    const float mean = red[0] * (1.0f / HID);
    __syncthreads();
    const float c = v - mean;
    red[d] = c * c; __syncthreads();
    for (int s = 64; s > 0; s >>= 1) { if (d < s) red[d] += red[d + s]; __syncthreads(); }
    const float var = red[0] * (1.0f / HID);
    out[(long)row * HID + d] = c * rsqrtf(var + 1e-5f) * g[d] + be[d];
}

// ---------------------------------------------------------------------------
// Brute-force KNN (K=8, self included, lower-index tie-break like lax.top_k),
// writing a bit-packed symmetric adjacency via atomicOr (to_bidirected union).
// ---------------------------------------------------------------------------
__global__ __launch_bounds__(256)
void knn_kernel(const float* __restrict__ cent, unsigned* __restrict__ adj,
                int Ncur, int total)
{
    const int t = blockIdx.x * blockDim.x + threadIdx.x;
    if (t >= total) return;
    const int b = t / Ncur, i = t % Ncur;
    const float xi = cent[((long)b * Ncur + i) * 2 + 0];
    const float yi = cent[((long)b * Ncur + i) * 2 + 1];

    float bd[KNN]; int bi[KNN];
#pragma unroll
    for (int q = 0; q < KNN; ++q) { bd[q] = 3.4e38f; bi[q] = 0x7fffffff; }

    for (int j = 0; j < Ncur; ++j) {
        const float dx = xi - cent[((long)b * Ncur + j) * 2 + 0];
        const float dy = yi - cent[((long)b * Ncur + j) * 2 + 1];
        const float d2 = dx * dx + dy * dy;
        if (d2 < bd[KNN - 1] || (d2 == bd[KNN - 1] && j < bi[KNN - 1])) {
            int p = KNN - 1;
            while (p > 0 && (d2 < bd[p - 1] || (d2 == bd[p - 1] && j < bi[p - 1]))) {
                bd[p] = bd[p - 1]; bi[p] = bi[p - 1]; --p;
            }
            bd[p] = d2; bi[p] = j;
        }
    }
#pragma unroll
    for (int q = 0; q < KNN; ++q) {
        const int j = bi[q];
        atomicOr(&adj[((long)b * Ncur + i) * ADJ_WORDS + (j >> 5)], 1u << (j & 31));
        atomicOr(&adj[((long)b * Ncur + j) * ADJ_WORDS + (i >> 5)], 1u << (i & 31));
    }
}

// ---------------------------------------------------------------------------
// Sparse mean aggregation: m[i,:] = (sum_{j in A_i} xn[j,:]) / deg[i].
// One block per node, 128 threads = feature lanes, bit-scan of adjacency row.
// ---------------------------------------------------------------------------
__global__ __launch_bounds__(128)
void aggregate_kernel(const unsigned* __restrict__ adj, const float* __restrict__ xn,
                      float* __restrict__ m, float* __restrict__ deg, int Ncur)
{
    const int node = blockIdx.x;                 // b*Ncur + i
    const int b    = node / Ncur;
    const int d    = threadIdx.x;
    const unsigned* row = adj + (long)node * ADJ_WORDS;
    const int words = Ncur >> 5;
    float acc = 0.0f; int cnt = 0;
    for (int w = 0; w < words; ++w) {
        unsigned bits = row[w];
        cnt += __popc(bits);
        while (bits) {
            const int j = (w << 5) + __ffs(bits) - 1;
            bits &= bits - 1;
            acc += xn[((long)b * Ncur + j) * HID + d];
        }
    }
    const float dv = (float)cnt;                 // >= 1 (self loop)
    m[(long)node * HID + d] = acc / dv;
    if (d == 0) deg[node] = dv;
}

// y[node] = dot(x[node,:], Wq)
__global__ __launch_bounds__(256)
void score_y_kernel(const float* __restrict__ x, const float* __restrict__ Wq,
                    float* __restrict__ y, int total)
{
    const int t = blockIdx.x * blockDim.x + threadIdx.x;
    if (t >= total) return;
    float acc = 0.0f;
    for (int d = 0; d < HID; ++d) acc += x[(long)t * HID + d] * Wq[d];
    y[t] = acc;
}

// s[i] = dinv[i] * sum_{j in A_i} dinv[j]*y[j] + bq
__global__ __launch_bounds__(256)
void score_agg_kernel(const unsigned* __restrict__ adj, const float* __restrict__ y,
                      const float* __restrict__ deg, const float* __restrict__ bq,
                      float* __restrict__ s, int Ncur, int total)
{
    const int t = blockIdx.x * blockDim.x + threadIdx.x;
    if (t >= total) return;
    const int b = t / Ncur;
    const unsigned* row = adj + (long)t * ADJ_WORDS;
    const int words = Ncur >> 5;
    float acc = 0.0f;
    for (int w = 0; w < words; ++w) {
        unsigned bits = row[w];
        while (bits) {
            const int j = (w << 5) + __ffs(bits) - 1;
            bits &= bits - 1;
            acc += rsqrtf(deg[(long)b * Ncur + j]) * y[(long)b * Ncur + j];
        }
    }
    s[t] = rsqrtf(deg[t]) * acc + bq[0];
}

// ---------------------------------------------------------------------------
// Stable top-k by exact rank (reproduces lax.top_k value-desc, index-asc tie
// ordering): rank(i) = #{j : s[j]>s[i] or (s[j]==s[i] and j<i)}.
// ---------------------------------------------------------------------------
__global__ __launch_bounds__(256)
void topk_kernel(const float* __restrict__ s, int* __restrict__ idx,
                 int Ncur, int kkeep)
{
    const int b = blockIdx.x;
    const float* sb = s + (long)b * Ncur;
    for (int i = threadIdx.x; i < Ncur; i += 256) {
        const float si = sb[i];
        int rank = 0;
        for (int j = 0; j < Ncur; ++j) {
            const float sj = sb[j];
            rank += (sj > si) || (sj == si && j < i);
        }
        if (rank < kkeep) idx[(long)b * kkeep + rank] = i;
    }
}

// Gather pooled nodes, gate features with tanh(score); also gather centroids.
__global__ __launch_bounds__(128)
void gather_kernel(const float* __restrict__ x, const float* __restrict__ c,
                   const float* __restrict__ s, const int* __restrict__ idx,
                   float* __restrict__ xo, float* __restrict__ co,
                   int Ncur, int kkeep)
{
    const int r = blockIdx.x;                    // b*kkeep + rr
    const int b = r / kkeep;
    const int j = idx[r];
    const float gate = tanhf(s[(long)b * Ncur + j]);
    const int d = threadIdx.x;
    xo[(long)r * HID + d] = x[((long)b * Ncur + j) * HID + d] * gate;
    if (d < 2) co[(long)r * 2 + d] = c[((long)b * Ncur + j) * 2 + d];
}

// picks[b, 0:128] += max over pooled rows, picks[b,128:256] += mean.
__global__ __launch_bounds__(256)
void readout_kernel(const float* __restrict__ xo, float* __restrict__ picks, int kkeep)
{
    const int t = blockIdx.x * blockDim.x + threadIdx.x;
    if (t >= BATCH * HID) return;
    const int b = t >> 7, d = t & 127;
    float mx = -3.4e38f, sum = 0.0f;
    for (int r = 0; r < kkeep; ++r) {
        const float v = xo[((long)b * kkeep + r) * HID + d];
        mx = fmaxf(mx, v); sum += v;
    }
    picks[b * (2 * HID) + d]       += mx;
    picks[b * (2 * HID) + HID + d] += sum / (float)kkeep;
}

// Final tiny MLP (M=8 rows: scalar kernels, WMMA not worth it here).
__global__ __launch_bounds__(256)
void final1_kernel(const float* __restrict__ picks, const float* __restrict__ Wf1,
                   const float* __restrict__ bf1, float* __restrict__ f)
{
    const int t = blockIdx.x * blockDim.x + threadIdx.x;   // 8*128
    if (t >= BATCH * HID) return;
    const int b = t >> 7, n = t & 127;
    float acc = bf1[n];
    for (int k = 0; k < 2 * HID; ++k) acc += picks[b * (2 * HID) + k] * Wf1[k * HID + n];
    f[t] = fmaxf(acc, 0.0f);
}

__global__ __launch_bounds__(256)
void final2_kernel(const float* __restrict__ f, const float* __restrict__ Wf2,
                   const float* __restrict__ bf2, float* __restrict__ out)
{
    const int t = blockIdx.x * blockDim.x + threadIdx.x;   // 8*32
    if (t >= BATCH * GRAPH_DIM) return;
    const int b = t >> 5, n = t & 31;
    float acc = bf2[n];
    for (int k = 0; k < HID; ++k) acc += f[b * HID + k] * Wf2[k * GRAPH_DIM + n];
    out[t] = fmaxf(acc, 0.0f);
}

// ---------------------------------------------------------------------------
// Host side
// ---------------------------------------------------------------------------
extern "C" void kernel_launch(void* const* d_in, const int* in_sizes, int n_in,
                              void* d_out, int out_size, void* d_ws, size_t ws_size,
                              hipStream_t stream)
{
    (void)in_sizes; (void)n_in; (void)out_size; (void)ws_size;
    const float* feats = (const float*)d_in[0];
    const float* cent  = (const float*)d_in[1];
    const float* Wp1 = (const float*)d_in[2];
    const float* bp1 = (const float*)d_in[3];
    const float* bn_g = (const float*)d_in[4];
    const float* bn_b = (const float*)d_in[5];
    const float* Wp2 = (const float*)d_in[6];
    const float* bp2 = (const float*)d_in[7];
    const float* g1  = (const float*)d_in[8];
    const float* be1 = (const float*)d_in[9];
    const float* g2  = (const float*)d_in[10];
    const float* be2 = (const float*)d_in[11];
    const float* Ws  = (const float*)d_in[12];
    const float* Wn  = (const float*)d_in[13];
    const float* bs  = (const float*)d_in[14];
    const float* Wg  = (const float*)d_in[15];
    const float* bg  = (const float*)d_in[16];
    const float* W1  = (const float*)d_in[17];
    const float* b1  = (const float*)d_in[18];
    const float* W2  = (const float*)d_in[19];
    const float* b2  = (const float*)d_in[20];
    const float* Wq  = (const float*)d_in[21];
    const float* bq  = (const float*)d_in[22];
    const float* Wf1 = (const float*)d_in[23];
    const float* bf1 = (const float*)d_in[24];
    const float* Wf2 = (const float*)d_in[25];
    const float* bf2 = (const float*)d_in[26];
    float* out = (float*)d_out;

    // ---- workspace carve-up (256B aligned) ----
    size_t off = 0;
    auto carve = [&](size_t bytes) -> char* {
        char* p = (char*)d_ws + off;
        off += (bytes + 255) & ~(size_t)255;
        return p;
    };
    const size_t xBytes = (size_t)BATCH * NMAX * HID * sizeof(float);      // 8 MB
    float*    bufX0 = (float*)carve(xBytes);
    float*    bufX1 = (float*)carve(xBytes);
    float*    bufXn = (float*)carve(xBytes);
    float*    bufM  = (float*)carve(xBytes);
    float*    bufH  = (float*)carve((size_t)BATCH * NMAX * NHEAD * HID * sizeof(float)); // 32 MB
    unsigned* adj   = (unsigned*)carve((size_t)BATCH * NMAX * ADJ_WORDS * sizeof(unsigned)); // 4 MB
    float*    deg   = (float*)carve((size_t)BATCH * NMAX * sizeof(float));
    float*    ybuf  = (float*)carve((size_t)BATCH * NMAX * sizeof(float));
    float*    sbuf  = (float*)carve((size_t)BATCH * NMAX * sizeof(float));
    int*      ibuf  = (int*)carve((size_t)BATCH * NMAX * sizeof(int));
    float*    centA = (float*)carve((size_t)BATCH * NMAX * 2 * sizeof(float));
    float*    centB = (float*)carve((size_t)BATCH * NMAX * 2 * sizeof(float));
    float*    picks = (float*)carve((size_t)BATCH * 2 * HID * sizeof(float));
    float*    fbuf  = (float*)carve((size_t)BATCH * HID * sizeof(float));
    // f16 packed weights (per-column contiguous)
    _Float16* wp1T = (_Float16*)carve((size_t)INFEAT * HID * sizeof(_Float16));
    _Float16* wp2T = (_Float16*)carve((size_t)HID * HID * sizeof(_Float16));
    _Float16* wsT[NLAYER]; _Float16* wnT[NLAYER]; _Float16* wgT[NLAYER];
    _Float16* w1T[NLAYER]; _Float16* w2T[NLAYER];
    for (int i = 0; i < NLAYER; ++i) {
        wsT[i] = (_Float16*)carve((size_t)HID * NHEAD * HID * sizeof(_Float16));
        wnT[i] = (_Float16*)carve((size_t)HID * NHEAD * HID * sizeof(_Float16));
        wgT[i] = (_Float16*)carve((size_t)NHEAD * HID * HID * sizeof(_Float16));
        w1T[i] = (_Float16*)carve((size_t)HID * HID * sizeof(_Float16));
        w2T[i] = (_Float16*)carve((size_t)HID * HID * sizeof(_Float16));
    }

    const float bn_mul = 1.0f / sqrtf(1.0f + 1e-5f);

    hipMemsetAsync(picks, 0, (size_t)BATCH * 2 * HID * sizeof(float), stream);

    // ---- pack all WMMA weights to f16 column-major once ----
    pack_w(Wp1, wp1T, INFEAT, HID, stream);
    pack_w(Wp2, wp2T, HID, HID, stream);
    for (int i = 0; i < NLAYER; ++i) {
        pack_w(Ws + (size_t)i * NHEAD * HID * HID, wsT[i], HID, NHEAD * HID, stream);
        pack_w(Wn + (size_t)i * NHEAD * HID * HID, wnT[i], HID, NHEAD * HID, stream);
        pack_w(Wg + (size_t)i * NHEAD * HID * HID, wgT[i], NHEAD * HID, HID, stream);
        pack_w(W1 + (size_t)i * HID * HID, w1T[i], HID, HID, stream);
        pack_w(W2 + (size_t)i * HID * HID, w2T[i], HID, HID, stream);
    }

    // ---- preconv: relu(bn(feats@Wp1+bp1)) @ Wp2 + bp2 ----
    {
        const int M = BATCH * NMAX;
        // K=512, bias + BN scale/shift + ReLU-out
        launch_gemm<INFEAT, false, true, true, true, false, false>(
            feats, nullptr, wp1T, nullptr, bp1, bn_g, bn_b, nullptr,
            bufXn, M, HID, bn_mul, stream);
        // K=128, bias only
        launch_gemm<HID, false, false, true, false, false, false>(
            bufXn, nullptr, wp2T, nullptr, bp2, nullptr, nullptr, nullptr,
            bufX0, M, HID, 1.0f, stream);
    }

    float* xs[2] = {bufX0, bufX1};
    float* cs[2] = {centA, centB};
    const float* c_cur = cent;
    int cur = 0;

    for (int i = 0; i < NLAYER; ++i) {
        const int Ncur  = NMAX >> i;
        const int rows  = BATCH * Ncur;
        const int kkeep = Ncur / 2;

        // dynamic KNN graph (bit-packed symmetric adjacency)
        hipMemsetAsync(adj, 0, (size_t)BATCH * NMAX * ADJ_WORDS * sizeof(unsigned), stream);
        knn_kernel<<<(rows + 255) / 256, 256, 0, stream>>>(c_cur, adj, Ncur, rows);

        // pre-LN -> mean SAGEConv
        layernorm_kernel<<<rows, 128, 0, stream>>>(xs[cur], g1 + i * HID, be1 + i * HID, bufXn);
        aggregate_kernel<<<rows, 128, 0, stream>>>(adj, bufXn, bufM, deg, Ncur);

        // h = relu(xn @ Ws_cat + m @ Wn_cat + bs)   (fused dual-A GEMM, Nout=512)
        launch_gemm<HID, false, true, true, false, false, true>(
            bufXn, bufM, wsT[i], wnT[i], bs + (size_t)i * NHEAD * HID,
            nullptr, nullptr, nullptr, bufH, rows, NHEAD * HID, 1.0f, stream);

        // x += h @ Wg + bg   (K=512, residual; h already rectified)
        launch_gemm<NHEAD * HID, false, false, true, false, true, false>(
            bufH, nullptr, wgT[i], nullptr, bg + (size_t)i * HID,
            nullptr, nullptr, xs[cur], xs[cur], rows, HID, 1.0f, stream);

        // FFN with residual
        layernorm_kernel<<<rows, 128, 0, stream>>>(xs[cur], g2 + i * HID, be2 + i * HID, bufXn);
        launch_gemm<HID, false, true, true, false, false, false>(
            bufXn, nullptr, w1T[i], nullptr, b1 + (size_t)i * HID,
            nullptr, nullptr, nullptr, bufM, rows, HID, 1.0f, stream);
        launch_gemm<HID, false, false, true, false, true, false>(
            bufM, nullptr, w2T[i], nullptr, b2 + (size_t)i * HID,
            nullptr, nullptr, xs[cur], xs[cur], rows, HID, 1.0f, stream);

        // SAGPool score -> top-k -> gate -> readout
        score_y_kernel<<<(rows + 255) / 256, 256, 0, stream>>>(xs[cur], Wq + (size_t)i * HID, ybuf, rows);
        score_agg_kernel<<<(rows + 255) / 256, 256, 0, stream>>>(adj, ybuf, deg, bq + i, sbuf, Ncur, rows);
        topk_kernel<<<BATCH, 256, 0, stream>>>(sbuf, ibuf, Ncur, kkeep);
        gather_kernel<<<BATCH * kkeep, 128, 0, stream>>>(xs[cur], c_cur, sbuf, ibuf,
                                                         xs[1 - cur], cs[i & 1], Ncur, kkeep);
        readout_kernel<<<(BATCH * HID + 255) / 256, 256, 0, stream>>>(xs[1 - cur], picks, kkeep);

        c_cur = cs[i & 1];
        cur = 1 - cur;
    }

    // final MLP
    final1_kernel<<<(BATCH * HID + 255) / 256, 256, 0, stream>>>(picks, Wf1, bf1, fbuf);
    final2_kernel<<<(BATCH * GRAPH_DIM + 255) / 256, 256, 0, stream>>>(fbuf, Wf2, bf2, out);
}
